// HydraChannelMixer_20512763806119
// MI455X (gfx1250) — matrix operand order repaired and checked by reading, not compiled, not run on hardware
//
#include <hip/hip_runtime.h>
#include <math.h>

// ---------------------------------------------------------------------------
// HydraChannelMixer for MI455X (gfx1250), wave32 + WMMA f32_16x16x32_f16.
// One wave per (b,p) tile of shape [C=32, D=128].  Memory-bound target:
// 512 MB @ 23.3 TB/s ~= 22 us; WMMA makes the 13.3 GFLOP of matmuls free.
// Tile load uses the Tensor Data Mover (one TENSOR_LOAD_TO_LDS per wave,
// with hardware LDS padding to the bank-friendly stride-132 layout).
// ---------------------------------------------------------------------------

typedef __attribute__((ext_vector_type(16))) _Float16 v16h;
typedef __attribute__((ext_vector_type(8)))  _Float16 v8h;
typedef __attribute__((ext_vector_type(4)))  _Float16 v4h;
typedef __attribute__((ext_vector_type(8)))  float    v8f;
typedef __attribute__((ext_vector_type(4)))  float    v4f;
typedef __attribute__((ext_vector_type(4)))  int      v4i;
typedef __attribute__((ext_vector_type(8)))  int      v8i;
typedef __attribute__((ext_vector_type(4)))  unsigned int v4u;

union AFrag { v16h v; v4h q[4]; };
union BFrag { v16h v; v8h o[2]; };
union CFrag { v8f  v; float f[8]; };

#if defined(__has_builtin)
#  if __has_builtin(__builtin_amdgcn_tensor_load_to_lds)
#    define HAS_TDM 1
#  endif
#  if __has_builtin(__builtin_amdgcn_global_load_async_to_lds_b128)
#    define HAS_ASYNC_LDS 1
#  endif
#endif

// async-to-LDS builtin takes (int4 AS1*, int4 AS3*, imm offset, imm cpol)
typedef __attribute__((address_space(1))) v4i g_v4i;
typedef __attribute__((address_space(3))) v4i l_v4i;

// fast transcendentals: v_exp_f32 / v_rcp_f32 / v_sqrt_f32 / v_rsq_f32
__device__ __forceinline__ float fast_sigmoid(float x) {
  return __builtin_amdgcn_rcpf(1.0f + __expf(-x));
}

// A-matrix (16x32 f16) fragment from an LDS row-major f16 buffer.
// lane<16 -> M=lane, K = {0..7,16..23}+Kt*32 ; lane>=16 -> K = {8..15,24..31}+Kt*32
__device__ __forceinline__ v16h load_A_lds(const _Float16* buf, int stride,
                                           int Mt, int Kt, int ln16, int g) {
  const _Float16* p = buf + (Mt * 16 + ln16) * stride + Kt * 32 + g * 8;
  AFrag a;
  a.q[0] = *(const v4h*)(p + 0);
  a.q[1] = *(const v4h*)(p + 4);
  a.q[2] = *(const v4h*)(p + 16);
  a.q[3] = *(const v4h*)(p + 20);
  return a.v;
}

// B-matrix (32x16 f16) fragment from a transposed (N-major) f16 weight in global.
// lane -> N = ln16 ; K = {0..7,16..23}+Kt*32 (g=0) or {8..15,24..31}+Kt*32 (g=1).
__device__ __forceinline__ v16h load_B_glb(const _Float16* __restrict__ wt, int kdim,
                                           int Nt, int Kt, int ln16, int g) {
  const _Float16* p = wt + (Nt * 16 + ln16) * kdim + Kt * 32 + g * 8;
  BFrag b;
  b.o[0] = *(const v8h*)(p);
  b.o[1] = *(const v8h*)(p + 16);
  return b.v;
}

// ---------------------------------------------------------------------------
// Workspace layout:
//   f16 region (halves):
//     [0,4096)     W_down^T [32][128]
//     [4096,5120)  W_q^T    [32][32]
//     [5120,6144)  W_k^T
//     [6144,7168)  W_v^T
//     [7168,8192)  W_cg^T
//     [8192,12288) W_up^T   [128][32]
//   f32 region (floats, byte offset 24576):
//     [0,8320)      Wg1^T [32][260]  (cols 257..259 zero-padded)
//     [8320,12416)  Wg2^T [128][32]
// ---------------------------------------------------------------------------
__global__ void hydra_prep_weights(const float* __restrict__ Wd, const float* __restrict__ Wq,
                                   const float* __restrict__ Wk, const float* __restrict__ Wv,
                                   const float* __restrict__ Wcg, const float* __restrict__ Wu,
                                   const float* __restrict__ Wg1, const float* __restrict__ Wg2,
                                   _Float16* __restrict__ wsh) {
  float* wsf = (float*)(wsh + 12288);
  int t = blockIdx.x * blockDim.x + threadIdx.x;
  if (t < 4096) {                         // W_down^T[r][d] = W_down[d][r]
    int r = t >> 7, d = t & 127;
    wsh[t] = (_Float16)Wd[d * 32 + r];
  } else if (t < 8192) {                  // four RxR transposes
    int u = t - 4096;
    int w = u >> 10;
    int jj = (u >> 5) & 31, ii = u & 31;  // wsh[.. + j*32 + i] = W[i*32 + j]
    const float* W = (w == 0) ? Wq : (w == 1) ? Wk : (w == 2) ? Wv : Wcg;
    wsh[t] = (_Float16)W[ii * 32 + jj];
  } else if (t < 12288) {                 // W_up^T[d][r] = W_up[r][d]
    int u = t - 8192;
    int d = u >> 5, r = u & 31;
    wsh[t] = (_Float16)Wu[r * 128 + d];
  } else if (t < 12288 + 8320) {          // Wg1^T[j][i] = Wg1[i][j], stride 260, zero pad
    int u = t - 12288;
    int j = u / 260, i = u % 260;
    wsf[u] = (i < 257) ? Wg1[i * 32 + j] : 0.0f;
  } else if (t < 12288 + 8320 + 4096) {   // Wg2^T[d][j] = Wg2[j][d]
    int u = t - (12288 + 8320);
    int d = u >> 5, j = u & 31;
    wsf[8320 + u] = Wg2[j * 128 + d];
  }
}

// ---------------------------------------------------------------------------
// Main kernel: one wave per bp = b*256 + p.
// ---------------------------------------------------------------------------
__global__ __launch_bounds__(32)
void hydra_main(const float* __restrict__ x,
                const float* __restrict__ gamma, const float* __restrict__ beta,
                const float* __restrict__ b_down, const float* __restrict__ b_cg,
                const float* __restrict__ b_up,
                const float* __restrict__ bg1, const float* __restrict__ bg2,
                const _Float16* __restrict__ wt,
                float* __restrict__ out) {
  const _Float16* WdT = wt;              // [32][128]
  const _Float16* WqT = wt + 4096;       // [32][32]
  const _Float16* WkT = wt + 5120;
  const _Float16* WvT = wt + 6144;
  const _Float16* WcT = wt + 7168;
  const _Float16* WuT = wt + 8192;       // [128][32]
  const float* Wg1T = (const float*)(wt + 12288);  // [32][260]
  const float* Wg2T = Wg1T + 8320;                 // [128][32]

  __shared__ __attribute__((aligned(16))) float    s_h[32 * 132];   // 16896 B
  __shared__ __attribute__((aligned(16))) _Float16 s_hl[32 * 36];   //  2304 B
  __shared__ __attribute__((aligned(16))) _Float16 s_am[32 * 36];   //  2304 B
  __shared__ __attribute__((aligned(16))) float    s_stats[64];     // mean/rstd per row
  __shared__ __attribute__((aligned(16))) float    s_gin[260];
  __shared__ __attribute__((aligned(16))) float    s_g1[32];
  __shared__ __attribute__((aligned(16))) float    s_gate[128];

  const int lane = threadIdx.x;          // 0..31, one wave
  const int ln16 = lane & 15;
  const int g    = lane >> 4;
  const int bp   = blockIdx.x;
  const int b    = bp >> 8;
  const int p    = bp & 255;

  // ---- load tile: h[c][d] = x[(b*32 + c)*256*128 + p*128 + d] --------------
  const float* xbase = x + ((size_t)(b * 32) * 256 + (size_t)p) * 128;
#if defined(HAS_TDM)
  {
    // Tensor DMA descriptor: 2D tile 128x32 f32, tensor row stride 32768 elts,
    // LDS padding: 4 DWORDs after every 128 DWORDs -> stride-132 rows.
    unsigned long long ga = (unsigned long long)(size_t)xbase;
    unsigned lds_off = (unsigned)(unsigned long long)(l_v4i*)&s_h[0];
    v4u g0 = { 1u,                                     // count=1, user descriptor
               lds_off,                                // lds_addr
               (unsigned)(ga & 0xffffffffu),           // global_addr[31:0]
               (unsigned)((ga >> 32) & 0x1ffffffu) | (2u << 30) };  // addr[56:32] | type=2
    v8i g1 = { (int)((2u << 16) | (1u << 20) | (6u << 22) | (3u << 25)),
               //  data_size=4B | pad_enable | pad_interval=128dw | pad_amount=4dw
               (int)(128u << 16),                      // tensor_dim0 = 128
               (int)(32u << 16),                       // tensor_dim1 = 32
               (int)(128u << 16),                      // tile_dim0 = 128
               (int)32,                                // tile_dim1 = 32
               (int)32768,                             // tensor_dim0_stride = 32768
               0, 0 };
    v4i g2 = {}, g3 = {};
#  if __clang_major__ >= 23
    v8i g4 = {};
    __builtin_amdgcn_tensor_load_to_lds(g0, g1, g2, g3, g4, 0);
#  else
    __builtin_amdgcn_tensor_load_to_lds(g0, g1, g2, g3, 0);
#  endif
    __builtin_amdgcn_s_wait_tensorcnt(0);
  }
#elif defined(HAS_ASYNC_LDS)
  #pragma unroll 8
  for (int c = 0; c < 32; ++c) {
    __builtin_amdgcn_global_load_async_to_lds_b128(
        (g_v4i*)(xbase + (size_t)c * 32768 + lane * 4),
        (l_v4i*)&s_h[c * 132 + lane * 4], 0, 0);
  }
#  if __has_builtin(__builtin_amdgcn_s_wait_asynccnt)
  __builtin_amdgcn_s_wait_asynccnt(0);
#  else
  asm volatile("s_wait_asynccnt 0" ::: "memory");
#  endif
#else
  #pragma unroll 4
  for (int c = 0; c < 32; ++c) {
    v4f t = *(const v4f*)(xbase + (size_t)c * 32768 + lane * 4);
    *(v4f*)&s_h[c * 132 + lane * 4] = t;
  }
#endif
  __syncthreads();

  // ---- LayerNorm stats, row c = lane ---------------------------------------
  {
    const float* row = &s_h[lane * 132];
    float sum = 0.f, ss = 0.f;
    #pragma unroll
    for (int j = 0; j < 32; ++j) {
      v4f t = *(const v4f*)(row + 4 * j);
      sum += t.x + t.y + t.z + t.w;
      ss  += t.x * t.x + t.y * t.y + t.z * t.z + t.w * t.w;
    }
    float mean = sum * (1.f / 128.f);
    float var  = ss * (1.f / 128.f) - mean * mean;
    s_stats[2 * lane]     = mean;
    s_stats[2 * lane + 1] = __builtin_amdgcn_rsqf(var + 1e-5f);
  }

  // ---- channel stats over C: d = 4*lane .. 4*lane+3 ------------------------
  {
    v4f sum = {}, ss = {}, sa = {};
    const float* col = &s_h[4 * lane];
    #pragma unroll 8
    for (int c = 0; c < 32; ++c) {
      v4f t = *(const v4f*)(col + c * 132);
      sum.x += t.x; sum.y += t.y; sum.z += t.z; sum.w += t.w;
      ss.x += t.x * t.x; ss.y += t.y * t.y; ss.z += t.z * t.z; ss.w += t.w * t.w;
      sa.x += fabsf(t.x); sa.y += fabsf(t.y); sa.z += fabsf(t.z); sa.w += fabsf(t.w);
    }
    v4f var, ma;
    var.x = (ss.x - sum.x * sum.x * (1.f / 32.f)) * (1.f / 31.f);
    var.y = (ss.y - sum.y * sum.y * (1.f / 32.f)) * (1.f / 31.f);
    var.z = (ss.z - sum.z * sum.z * (1.f / 32.f)) * (1.f / 31.f);
    var.w = (ss.w - sum.w * sum.w * (1.f / 32.f)) * (1.f / 31.f);
    ma.x = sa.x * (1.f / 32.f); ma.y = sa.y * (1.f / 32.f);
    ma.z = sa.z * (1.f / 32.f); ma.w = sa.w * (1.f / 32.f);
    *(v4f*)&s_gin[4 * lane]       = var;
    *(v4f*)&s_gin[128 + 4 * lane] = ma;
  }
  if (lane < 4) s_gin[256 + lane] = (lane == 0) ? 0.50171665944f : 0.0f; // log(32)/log(1000)
  __syncthreads();

  // ---- adaptive gate MLP ---------------------------------------------------
  {
    const float* wrow = Wg1T + lane * 260;   // contiguous, 16B aligned
    float acc = bg1[lane];
    #pragma unroll 5
    for (int i4 = 0; i4 < 65; ++i4) {
      v4f w  = *(const v4f*)(wrow + 4 * i4);
      v4f gi = *(const v4f*)(&s_gin[4 * i4]);
      acc += w.x * gi.x + w.y * gi.y + w.z * gi.z + w.w * gi.w;
    }
    s_g1[lane] = 0.5f * acc * (1.f + erff(acc * 0.70710678118f));  // exact gelu
  }
  __syncthreads();
  #pragma unroll
  for (int kk = 0; kk < 4; ++kk) {
    int d = lane + 32 * kk;
    const float* wrow = Wg2T + d * 32;
    float acc = bg2[d];
    #pragma unroll
    for (int j4 = 0; j4 < 8; ++j4) {
      v4f w  = *(const v4f*)(wrow + 4 * j4);
      v4f gv = *(const v4f*)(&s_g1[4 * j4]);
      acc += w.x * gv.x + w.y * gv.y + w.z * gv.z + w.w * gv.w;
    }
    s_gate[d] = fast_sigmoid(acc);
  }

  // ---- down-projection with fused LayerNorm (WMMA, K=128) -----------------
  #pragma unroll
  for (int Mt = 0; Mt < 2; ++Mt) {
    const int M = Mt * 16 + ln16;
    const float rstd = s_stats[2 * M + 1];
    const float c0   = -s_stats[2 * M] * rstd;   // -mean*rstd
    const float* hrow = &s_h[M * 132];
    v16h A[4];
    #pragma unroll
    for (int Kt = 0; Kt < 4; ++Kt) {
      AFrag a;
      #pragma unroll
      for (int c4 = 0; c4 < 4; ++c4) {
        const int off = (c4 & 1) * 4 + (c4 >> 1) * 16;  // 0,4,16,20
        const int d0  = Kt * 32 + g * 8 + off;
        v4f t  = *(const v4f*)(hrow + d0);
        v4f gm = *(const v4f*)(gamma + d0);
        v4f bt = *(const v4f*)(beta + d0);
        v4h o;
        o.x = (_Float16)fmaf(fmaf(t.x, rstd, c0), gm.x, bt.x);
        o.y = (_Float16)fmaf(fmaf(t.y, rstd, c0), gm.y, bt.y);
        o.z = (_Float16)fmaf(fmaf(t.z, rstd, c0), gm.z, bt.z);
        o.w = (_Float16)fmaf(fmaf(t.w, rstd, c0), gm.w, bt.w);
        a.q[c4] = o;
      }
      A[Kt] = a.v;
    }
    #pragma unroll
    for (int Nt = 0; Nt < 2; ++Nt) {
      v8f acc = {};
      #pragma unroll
      for (int Kt = 0; Kt < 4; ++Kt) {
        v16h B = load_B_glb(WdT, 128, Nt, Kt, ln16, g);
        acc = __builtin_amdgcn_wmma_f32_16x16x32_f16(false, A[Kt], false, B,
                                                     (short)0, acc, false, false);
      }
      float bias = b_down[Nt * 16 + ln16];
      CFrag cf; cf.v = acc;
      _Float16* pp = &s_hl[(Mt * 16 + g * 8) * 36 + Nt * 16 + ln16];
      #pragma unroll
      for (int r = 0; r < 8; ++r) pp[r * 36] = (_Float16)(cf.f[r] + bias);
    }
  }
  __syncthreads();

  // ---- Q,K,V,CG entirely in registers (WMMA C-layout + shfl reductions) ----
  {
    const v8f zero = {};
    v16h Alo0 = load_A_lds(s_hl, 36, 0, 0, ln16, g);
    v16h Alo1 = load_A_lds(s_hl, 36, 1, 0, ln16, g);

    // K and V tiles
    CFrag Kf[2][2], Vf[2][2];
    {
      v16h Bk0 = load_B_glb(WkT, 32, 0, 0, ln16, g);
      v16h Bk1 = load_B_glb(WkT, 32, 1, 0, ln16, g);
      v16h Bv0 = load_B_glb(WvT, 32, 0, 0, ln16, g);
      v16h Bv1 = load_B_glb(WvT, 32, 1, 0, ln16, g);
      Kf[0][0].v = __builtin_amdgcn_wmma_f32_16x16x32_f16(false, Alo0, false, Bk0, (short)0, zero, false, false);
      Kf[0][1].v = __builtin_amdgcn_wmma_f32_16x16x32_f16(false, Alo0, false, Bk1, (short)0, zero, false, false);
      Kf[1][0].v = __builtin_amdgcn_wmma_f32_16x16x32_f16(false, Alo1, false, Bk0, (short)0, zero, false, false);
      Kf[1][1].v = __builtin_amdgcn_wmma_f32_16x16x32_f16(false, Alo1, false, Bk1, (short)0, zero, false, false);
      Vf[0][0].v = __builtin_amdgcn_wmma_f32_16x16x32_f16(false, Alo0, false, Bv0, (short)0, zero, false, false);
      Vf[0][1].v = __builtin_amdgcn_wmma_f32_16x16x32_f16(false, Alo0, false, Bv1, (short)0, zero, false, false);
      Vf[1][0].v = __builtin_amdgcn_wmma_f32_16x16x32_f16(false, Alo1, false, Bv0, (short)0, zero, false, false);
      Vf[1][1].v = __builtin_amdgcn_wmma_f32_16x16x32_f16(false, Alo1, false, Bv1, (short)0, zero, false, false);
    }

    // row l2norm of K: reduce over N (lanes ln16, both Nt tiles)
    float kinv[2][8];
    #pragma unroll
    for (int Mt = 0; Mt < 2; ++Mt)
      #pragma unroll
      for (int r = 0; r < 8; ++r) {
        float s = Kf[Mt][0].f[r] * Kf[Mt][0].f[r] + Kf[Mt][1].f[r] * Kf[Mt][1].f[r];
        s += __shfl_xor(s, 1);  s += __shfl_xor(s, 2);
        s += __shfl_xor(s, 4);  s += __shfl_xor(s, 8);
        kinv[Mt][r] = __builtin_amdgcn_rcpf(fmaxf(__builtin_amdgcn_sqrtf(s), 1e-12f));
      }

    // global_feat[N] = sum over rows of khat * V  (per-lane N = Nt*16+ln16)
    float gf0 = 0.f, gf1 = 0.f;
    #pragma unroll
    for (int Mt = 0; Mt < 2; ++Mt)
      #pragma unroll
      for (int r = 0; r < 8; ++r) {
        float kh = kinv[Mt][r];
        gf0 += Kf[Mt][0].f[r] * kh * Vf[Mt][0].f[r];
        gf1 += Kf[Mt][1].f[r] * kh * Vf[Mt][1].f[r];
      }
    gf0 += __shfl_xor(gf0, 16);
    gf1 += __shfl_xor(gf1, 16);

    // Q and CG tiles
    CFrag Qf[2][2], Cg[2][2];
    {
      v16h Bq0 = load_B_glb(WqT, 32, 0, 0, ln16, g);
      v16h Bq1 = load_B_glb(WqT, 32, 1, 0, ln16, g);
      v16h Bc0 = load_B_glb(WcT, 32, 0, 0, ln16, g);
      v16h Bc1 = load_B_glb(WcT, 32, 1, 0, ln16, g);
      Qf[0][0].v = __builtin_amdgcn_wmma_f32_16x16x32_f16(false, Alo0, false, Bq0, (short)0, zero, false, false);
      Qf[0][1].v = __builtin_amdgcn_wmma_f32_16x16x32_f16(false, Alo0, false, Bq1, (short)0, zero, false, false);
      Qf[1][0].v = __builtin_amdgcn_wmma_f32_16x16x32_f16(false, Alo1, false, Bq0, (short)0, zero, false, false);
      Qf[1][1].v = __builtin_amdgcn_wmma_f32_16x16x32_f16(false, Alo1, false, Bq1, (short)0, zero, false, false);
      Cg[0][0].v = __builtin_amdgcn_wmma_f32_16x16x32_f16(false, Alo0, false, Bc0, (short)0, zero, false, false);
      Cg[0][1].v = __builtin_amdgcn_wmma_f32_16x16x32_f16(false, Alo0, false, Bc1, (short)0, zero, false, false);
      Cg[1][0].v = __builtin_amdgcn_wmma_f32_16x16x32_f16(false, Alo1, false, Bc0, (short)0, zero, false, false);
      Cg[1][1].v = __builtin_amdgcn_wmma_f32_16x16x32_f16(false, Alo1, false, Bc1, (short)0, zero, false, false);
    }

    float qinv[2][8];
    #pragma unroll
    for (int Mt = 0; Mt < 2; ++Mt)
      #pragma unroll
      for (int r = 0; r < 8; ++r) {
        float s = Qf[Mt][0].f[r] * Qf[Mt][0].f[r] + Qf[Mt][1].f[r] * Qf[Mt][1].f[r];
        s += __shfl_xor(s, 1);  s += __shfl_xor(s, 2);
        s += __shfl_xor(s, 4);  s += __shfl_xor(s, 8);
        qinv[Mt][r] = __builtin_amdgcn_rcpf(fmaxf(__builtin_amdgcn_sqrtf(s), 1e-12f));
      }

    // h_attn * content-gate -> s_am (f16 A-layout source for up-proj)
    const float bcg0 = b_cg[ln16];
    const float bcg1 = b_cg[16 + ln16];
    #pragma unroll
    for (int Mt = 0; Mt < 2; ++Mt)
      #pragma unroll
      for (int r = 0; r < 8; ++r) {
        int row = Mt * 16 + g * 8 + r;
        float qh = qinv[Mt][r];
        float a0 = Qf[Mt][0].f[r] * qh * gf0 * fast_sigmoid(Cg[Mt][0].f[r] + bcg0);
        float a1 = Qf[Mt][1].f[r] * qh * gf1 * fast_sigmoid(Cg[Mt][1].f[r] + bcg1);
        s_am[row * 36 + ln16]      = (_Float16)a0;
        s_am[row * 36 + 16 + ln16] = (_Float16)a1;
      }
  }
  __syncthreads();

  // ---- up-projection + gate + residual, fused store (WMMA, K=32) -----------
  const size_t outb = ((size_t)(b * 32) * 256 + (size_t)p) * 128;
  {
    v16h Aup0 = load_A_lds(s_am, 36, 0, 0, ln16, g);
    v16h Aup1 = load_A_lds(s_am, 36, 1, 0, ln16, g);
    #pragma unroll
    for (int Mt = 0; Mt < 2; ++Mt) {
      #pragma unroll
      for (int Nt = 0; Nt < 8; ++Nt) {
        v16h B = load_B_glb(WuT, 32, Nt, 0, ln16, g);
        v8f z = {};
        v8f acc = __builtin_amdgcn_wmma_f32_16x16x32_f16(false, Mt ? Aup1 : Aup0, false, B,
                                                         (short)0, z, false, false);
        int d    = Nt * 16 + ln16;
        float gv = s_gate[d];
        float bu = b_up[d];
        CFrag cf; cf.v = acc;
        #pragma unroll
        for (int r = 0; r < 8; ++r) {
          int c = Mt * 16 + g * 8 + r;
          float val = s_h[c * 132 + d] + gv * (cf.f[r] + bu);
          out[outb + (size_t)c * 32768 + d] = val;
        }
      }
    }
  }
}

// ---------------------------------------------------------------------------
extern "C" void kernel_launch(void* const* d_in, const int* in_sizes, int n_in,
                              void* d_out, int out_size, void* d_ws, size_t ws_size,
                              hipStream_t stream) {
  const float* x      = (const float*)d_in[0];
  const float* gamma  = (const float*)d_in[1];
  const float* beta   = (const float*)d_in[2];
  const float* W_down = (const float*)d_in[3];
  const float* b_down = (const float*)d_in[4];
  const float* W_q    = (const float*)d_in[5];
  const float* W_k    = (const float*)d_in[6];
  const float* W_v    = (const float*)d_in[7];
  const float* W_cg   = (const float*)d_in[8];
  const float* b_cg   = (const float*)d_in[9];
  const float* W_up   = (const float*)d_in[10];
  const float* b_up   = (const float*)d_in[11];
  const float* Wg1    = (const float*)d_in[12];
  const float* bg1    = (const float*)d_in[13];
  const float* Wg2    = (const float*)d_in[14];
  const float* bg2    = (const float*)d_in[15];
  _Float16* wt = (_Float16*)d_ws;   // ~73 KB: f16 transposed weights + f32 gate weights

  hipLaunchKernelGGL(hydra_prep_weights, dim3(97), dim3(256), 0, stream,
                     W_down, W_q, W_k, W_v, W_cg, W_up, Wg1, Wg2, wt);
  hipLaunchKernelGGL(hydra_main, dim3(64 * 256), dim3(32), 0, stream,
                     x, gamma, beta, b_down, b_cg, b_up,
                     bg1, bg2, wt, (float*)d_out);
}
